// BashTransformer_21122649162394
// MI455X (gfx1250) — matrix operand order repaired and checked by reading, not compiled, not run on hardware
//
#include <hip/hip_runtime.h>
#include <hip/hip_bf16.h>
#include <math.h>

#define B_    4
#define L_    1024
#define BL_   4096
#define HID_  512
#define NH_   8
#define HD_   64
#define SH_   8
#define SD_   32
#define SDIM_ 256
#define NL_   6
#define FFN_  1536
#define VOC_  63

typedef __bf16 bf16x16 __attribute__((ext_vector_type(16)));
typedef float f32x8    __attribute__((ext_vector_type(8)));
typedef unsigned int v4u __attribute__((ext_vector_type(4)));
typedef int v8i __attribute__((ext_vector_type(8)));
typedef int v4i __attribute__((ext_vector_type(4)));

#if defined(__has_builtin)
#if __has_builtin(__builtin_amdgcn_tensor_load_to_lds) && \
    __has_builtin(__builtin_amdgcn_s_wait_tensorcnt)
#define USE_TDM 1
#endif
#endif

union FragU { bf16x16 v; uint4 q[2]; };

__device__ __forceinline__ unsigned short f2bf(float f) {
    unsigned int u = __float_as_uint(f);
    unsigned int r = u + 0x7fffu + ((u >> 16) & 1u);   // round-to-nearest-even
    return (unsigned short)(r >> 16);
}

// A fragment (16x32 bf16): lane<16 holds K {0..7,16..23}, lane>=16 {8..15,24..31}.
__device__ __forceinline__ bf16x16 ld_frag_a(const unsigned short* rowptr, int base1) {
    FragU u;
    u.q[0] = *(const uint4*)(rowptr + base1);
    u.q[1] = *(const uint4*)(rowptr + base1 + 16);
    return u.v;
}
// B fragment (32x16 bf16): lane<16 holds K {0..7,16..23}, lane>=16 {8..15,24..31}, N=lane%16.
__device__ __forceinline__ bf16x16 ld_frag_b(const unsigned short* rowptr, int hf) {
    FragU u;
    const uint4* p = (const uint4*)(rowptr + hf * 8);
    u.q[0] = p[0];
    u.q[1] = p[2];
    return u.v;
}
__device__ __forceinline__ f32x8 wmma_bf16(bf16x16 a, bf16x16 b, f32x8 c) {
    return __builtin_amdgcn_wmma_f32_16x16x32_bf16(false, a, false, b, (short)0, c,
                                                   false, false);
}
__device__ __forceinline__ float sigmoidf_(float x) { return 1.0f / (1.0f + __expf(-x)); }

#ifdef USE_TDM
// Issue a TDM load of a 2D bf16 tile (tile_dim0=32 K-elems, tile_dim1=64 rows) into
// LDS with row padding 64B data + 16B pad (matches LDS stride of 40 ushorts).
// D# layout per CDNA5 ISA ch.8 (group0 128b, group1 256b).
// clang-23 toolchain: 6-arg builtin (g0 v4u, g1 v8i, v4i, v4i, v8i, i32 cpol).
__device__ __forceinline__ void tdm_issue(unsigned lds_off, const unsigned short* gptr,
                                          int rem_n, int rem_k, int K) {
    unsigned long long ga = (unsigned long long)gptr;
    v4u g0;
    g0.x = 1u;                                              // count=1, user mode
    g0.y = lds_off;                                         // lds_addr (bytes)
    g0.z = (unsigned)(ga & 0xffffffffu);                    // global_addr[31:0]
    g0.w = (unsigned)((ga >> 32) & 0x1ffffffu) | (2u << 30); // addr[56:32], type=2
    unsigned td0 = (unsigned)rem_k;                         // tensor_dim0 (elements)
    unsigned td1 = (unsigned)rem_n;                         // tensor_dim1 (rows)
    unsigned long long st0 = (unsigned long long)K;         // dim0 stride (elements)
    v8i g1;
    g1[0] = (int)((1u << 16)            // data_size = 2 bytes
                | (1u << 20)            // pad_enable
                | (3u << 22)            // pad_interval: 16 DWORDs (=64B row)
                | (3u << 25));          // pad_amount: 4 DWORDs (=16B pad)
    g1[1] = (int)((td0 & 0xffffu) << 16);                   // tensor_dim0[15:0]
    g1[2] = (int)((td0 >> 16) | ((td1 & 0xffffu) << 16));   // dim0[31:16], dim1[15:0]
    g1[3] = (int)((td1 >> 16) | (32u << 16));               // dim1[31:16], tile_dim0=32
    g1[4] = (int)(64u);                                     // tile_dim1=64, tile_dim2=0
    g1[5] = (int)(st0 & 0xffffffffu);                       // dim0_stride[31:0]
    g1[6] = (int)((st0 >> 32) & 0xffffu);                   // dim0_stride[47:32]
    g1[7] = 0;
    v4i gz4 = {0, 0, 0, 0};
    v8i gz8 = {0, 0, 0, 0, 0, 0, 0, 0};
    __builtin_amdgcn_tensor_load_to_lds(g0, g1, gz4, gz4, gz8, 0);
}
#endif

// -------------------------------------------------------------------------
// GEMM:  C[M,N] = A[M,K] @ Bw[N,K]^T  (A, Bw bf16; f32 accumulate)
//   mode 0: store   mode 1: sigmoid(acc + bias[n])
//   mode 2: acc + extra[m,n]   mode 3: silu(acc)*extra[m,n]
// Optional f32 C and bf16 Cbf outputs. M multiple of 128, K multiple of 32.
// B tile streamed by the Tensor Data Mover (double buffered).
// -------------------------------------------------------------------------
#define LST_ 40
__device__ __forceinline__ void gemm_store(float* C, unsigned short* Cbf,
                                           const float* extra, int mode,
                                           int m, int n, int N, float v) {
    if (n >= N) return;
    size_t off = (size_t)m * N + n;
    if (mode == 1)      v = sigmoidf_(v + extra[n]);
    else if (mode == 2) v = v + extra[off];
    else if (mode == 3) v = v * sigmoidf_(v) * extra[off];
    if (C)   C[off] = v;
    if (Cbf) Cbf[off] = f2bf(v);
}

__global__ __launch_bounds__(256) void gemm_wmma(const unsigned short* __restrict__ A,
                                                 const unsigned short* __restrict__ Bw,
                                                 float* __restrict__ C,
                                                 unsigned short* __restrict__ Cbf,
                                                 const float* __restrict__ extra,
                                                 int M, int N, int K, int mode) {
    __shared__ __align__(16) unsigned short As[128 * LST_];
    __shared__ __align__(16) unsigned short Bs[2][64 * LST_];
    const int tid = threadIdx.x;
    const int wave = tid >> 5, lane = tid & 31;
    const int hf = lane >> 4, lm = lane & 15;
    const int m0 = blockIdx.y * 128, n0 = blockIdx.x * 64;
    const int rb = (wave >> 1) * 32;   // wave 32-row block
    const int cb = (wave & 1) * 32;    // wave 32-col block
    f32x8 a00 = {}, a01 = {}, a10 = {}, a11 = {};
    const int nchunks = K >> 5;
    const unsigned short* btile = Bw + (size_t)n0 * K;

#ifdef USE_TDM
    if (tid < 32) {
        unsigned lds0 = (unsigned)(unsigned long long)(void*)&Bs[0][0];
        tdm_issue(lds0, btile, N - n0, K, K);
    }
#endif
    const int sr = tid >> 1;           // 0..127
    const int sc = (tid & 1) * 16;     // 0 / 16

    for (int ic = 0; ic < nchunks; ++ic) {
        const int k0 = ic << 5;
#ifdef USE_TDM
        if (tid < 32) {
            if (ic + 1 < nchunks) {    // prefetch next B tile into other buffer
                unsigned lds1 =
                    (unsigned)(unsigned long long)(void*)&Bs[(ic + 1) & 1][0];
                tdm_issue(lds1, btile + k0 + 32, N - n0, K - k0 - 32, K);
                __builtin_amdgcn_s_wait_tensorcnt(1);
            } else {
                __builtin_amdgcn_s_wait_tensorcnt(0);
            }
        }
        const unsigned short* bp = &Bs[ic & 1][0];
#else
        {   // manual B staging fallback: 64x32, 8 elems/thread
            int r = tid >> 2, c = (tid & 3) * 8;
            uint4* lb = (uint4*)&Bs[0][r * LST_ + c];
            if (n0 + r < N) {
                *lb = *(const uint4*)(btile + (size_t)r * K + k0 + c);
            } else {
                uint4 z = {0, 0, 0, 0};
                *lb = z;
            }
        }
        const unsigned short* bp = &Bs[0][0];
#endif
        {   // stage A: 128x32 bf16 copy, 16 elems/thread
            const uint4* ga = (const uint4*)(A + (size_t)(m0 + sr) * K + k0 + sc);
            uint4* la = (uint4*)&As[sr * LST_ + sc];
            la[0] = ga[0];
            la[1] = ga[1];
        }
        if (k0 + 32 < K) __builtin_prefetch(A + (size_t)(m0 + sr) * K + k0 + 32 + sc, 0, 1);
        __syncthreads();

        bf16x16 fa0 = ld_frag_a(&As[(rb + lm) * LST_], hf * 8);
        bf16x16 fa1 = ld_frag_a(&As[(rb + 16 + lm) * LST_], hf * 8);
        bf16x16 fb0 = ld_frag_b(bp + (cb + lm) * LST_, hf);
        bf16x16 fb1 = ld_frag_b(bp + (cb + 16 + lm) * LST_, hf);
        a00 = wmma_bf16(fa0, fb0, a00);
        a01 = wmma_bf16(fa0, fb1, a01);
        a10 = wmma_bf16(fa1, fb0, a10);
        a11 = wmma_bf16(fa1, fb1, a11);
        __syncthreads();
    }

#pragma unroll
    for (int i = 0; i < 8; ++i) {
        int mA = m0 + rb + i + hf * 8;
        int mB = mA + 16;
        int nA = n0 + cb + lm;
        int nB = nA + 16;
        gemm_store(C, Cbf, extra, mode, mA, nA, N, a00[i]);
        gemm_store(C, Cbf, extra, mode, mA, nB, N, a01[i]);
        gemm_store(C, Cbf, extra, mode, mB, nA, N, a10[i]);
        gemm_store(C, Cbf, extra, mode, mB, nB, N, a11[i]);
    }
}

// -------------------------------------------------------------------------
// Flash attention, causal.  grid=(L/64, B*NH), block=128 (4 waves).
// q/k/v layout: [b*L + l][h*64 + d] fp32; output written bf16 (GEMM A input).
// -------------------------------------------------------------------------
#define LKV_ 72
__global__ __launch_bounds__(128) void attn_flash(const float* __restrict__ Q,
                                                  const float* __restrict__ Kx,
                                                  const float* __restrict__ V,
                                                  unsigned short* __restrict__ Obf) {
    __shared__ __align__(16) unsigned short Qs[64 * LKV_];
    __shared__ __align__(16) unsigned short Ks[32 * LKV_];
    __shared__ __align__(16) unsigned short Vt[64 * 40];   // transposed [hd][key]
    __shared__ __align__(16) unsigned short Ps[4][16 * 40];
    const int tid = threadIdx.x, wave = tid >> 5, lane = tid & 31;
    const int hf = lane >> 4, lm = lane & 15;
    const int b = blockIdx.y / NH_, h = blockIdx.y % NH_;
    const int q0 = blockIdx.x * 64;

    for (int e = tid; e < 64 * 64; e += 128) {
        int r = e >> 6, c = e & 63;
        Qs[r * LKV_ + c] =
            f2bf(Q[(size_t)(b * L_ + q0 + r) * HID_ + h * HD_ + c] * 0.125f);
    }
    __syncthreads();

    const int arow = wave * 16 + lm;
    bf16x16 qa0 = ld_frag_a(&Qs[arow * LKV_ + 0], hf * 8);
    bf16x16 qa1 = ld_frag_a(&Qs[arow * LKV_ + 32], hf * 8);

    float mrun[8], lrun[8];
#pragma unroll
    for (int i = 0; i < 8; ++i) { mrun[i] = -1e30f; lrun[i] = 0.0f; }
    f32x8 o0 = {}, o1 = {}, o2 = {}, o3 = {};

    for (int kb = 0; kb < q0 + 64; kb += 32) {
        for (int e = tid; e < 32 * 64; e += 128) {
            int r = e >> 6, c = e & 63;
            size_t gi = (size_t)(b * L_ + kb + r) * HID_ + h * HD_ + c;
            Ks[r * LKV_ + c] = f2bf(Kx[gi]);
            Vt[c * 40 + r]   = f2bf(V[gi]);
        }
        __syncthreads();

        f32x8 s0 = {}, s1 = {};
        {
            bf16x16 bk = ld_frag_b(&Ks[lm * LKV_ + 0], hf);
            s0 = wmma_bf16(qa0, bk, s0);
            bk = ld_frag_b(&Ks[lm * LKV_ + 32], hf);
            s0 = wmma_bf16(qa1, bk, s0);
            bk = ld_frag_b(&Ks[(16 + lm) * LKV_ + 0], hf);
            s1 = wmma_bf16(qa0, bk, s1);
            bk = ld_frag_b(&Ks[(16 + lm) * LKV_ + 32], hf);
            s1 = wmma_bf16(qa1, bk, s1);
        }

#pragma unroll
        for (int i = 0; i < 8; ++i) {
            int m  = q0 + wave * 16 + i + hf * 8;
            int nA = kb + lm, nB = kb + 16 + lm;
            float a = (nA <= m) ? s0[i] : -1e30f;
            float c = (nB <= m) ? s1[i] : -1e30f;
            float mx = fmaxf(a, c);
            mx = fmaxf(mx, __shfl_xor(mx, 1));
            mx = fmaxf(mx, __shfl_xor(mx, 2));
            mx = fmaxf(mx, __shfl_xor(mx, 4));
            mx = fmaxf(mx, __shfl_xor(mx, 8));
            float mn = fmaxf(mrun[i], mx);
            float sc = __expf(mrun[i] - mn);
            mrun[i] = mn;
            float p0 = __expf(a - mn), p1 = __expf(c - mn);
            float rs = p0 + p1;
            rs += __shfl_xor(rs, 1);
            rs += __shfl_xor(rs, 2);
            rs += __shfl_xor(rs, 4);
            rs += __shfl_xor(rs, 8);
            lrun[i] = lrun[i] * sc + rs;
            o0[i] *= sc; o1[i] *= sc; o2[i] *= sc; o3[i] *= sc;
            int pr = i + hf * 8;
            Ps[wave][pr * 40 + lm]      = f2bf(p0);
            Ps[wave][pr * 40 + 16 + lm] = f2bf(p1);
        }
        asm volatile("s_wait_dscnt 0x0" ::: "memory");   // intra-wave LDS RAW

        bf16x16 pa = ld_frag_a(&Ps[wave][lm * 40], hf * 8);
        bf16x16 bv = ld_frag_b(&Vt[lm * 40], hf);
        o0 = wmma_bf16(pa, bv, o0);
        bv = ld_frag_b(&Vt[(16 + lm) * 40], hf);
        o1 = wmma_bf16(pa, bv, o1);
        bv = ld_frag_b(&Vt[(32 + lm) * 40], hf);
        o2 = wmma_bf16(pa, bv, o2);
        bv = ld_frag_b(&Vt[(48 + lm) * 40], hf);
        o3 = wmma_bf16(pa, bv, o3);
        __syncthreads();
    }

#pragma unroll
    for (int i = 0; i < 8; ++i) {
        int m = q0 + wave * 16 + i + hf * 8;
        float inv = 1.0f / lrun[i];
        size_t rbs = (size_t)(b * L_ + m) * HID_ + h * HD_;
        Obf[rbs + lm]      = f2bf(o0[i] * inv);
        Obf[rbs + 16 + lm] = f2bf(o1[i] * inv);
        Obf[rbs + 32 + lm] = f2bf(o2[i] * inv);
        Obf[rbs + 48 + lm] = f2bf(o3[i] * inv);
    }
}

// -------------------------------------------------------------------------
// Gated delta-rule scan.  grid = B*SH blocks of 32 threads; lane i owns row i of S.
// -------------------------------------------------------------------------
__global__ __launch_bounds__(32) void delta_scan(const float* __restrict__ kb,
                                                 const float* __restrict__ vb,
                                                 const float* __restrict__ qb,
                                                 const float* __restrict__ bb,
                                                 const float* __restrict__ ab,
                                                 const float* __restrict__ Sin,
                                                 float* __restrict__ Sout,
                                                 float* __restrict__ ctx) {
    const int b = blockIdx.x / SH_, hh = blockIdx.x % SH_;
    const int lane = threadIdx.x;
    float S[SD_];
    const size_t sbase = ((size_t)(b * SH_ + hh) * SD_ + lane) * SD_;
#pragma unroll
    for (int j = 0; j < SD_; ++j) S[j] = Sin[sbase + j];
    __shared__ float kk[SD_], qq[SD_];

    for (int l = 0; l < L_; ++l) {
        size_t idx = (size_t)(b * L_ + l) * SDIM_ + hh * SD_ + lane;
        float kv = kb[idx];
        float ss = kv * kv;
        ss += __shfl_xor(ss, 1);  ss += __shfl_xor(ss, 2);
        ss += __shfl_xor(ss, 4);  ss += __shfl_xor(ss, 8);
        ss += __shfl_xor(ss, 16);
        kv = kv / fmaxf(sqrtf(ss), 1e-12f);
        kk[lane] = kv;
        qq[lane] = qb[idx];
        __syncthreads();
        float a_i = ab[idx], b_i = bb[idx], v_i = vb[idx];
        float pred = 0.0f;
#pragma unroll
        for (int j = 0; j < SD_; ++j) { S[j] *= a_i; pred += S[j] * kk[j]; }
        float coef = b_i * (v_i - pred);
        float rd = 0.0f;
#pragma unroll
        for (int j = 0; j < SD_; ++j) { S[j] += coef * kk[j]; rd += S[j] * qq[j]; }
        ctx[idx] = rd;
        __syncthreads();
    }
#pragma unroll
    for (int j = 0; j < SD_; ++j) Sout[sbase + j] = S[j];
}

// -------------------------------------------------------------------------
// Elementwise / norm helpers
// -------------------------------------------------------------------------
__global__ __launch_bounds__(256) void cvt_bf16_k(const float* __restrict__ s,
                                                  unsigned short* __restrict__ d,
                                                  int n) {
    int i = (blockIdx.x * 256 + threadIdx.x) * 4;
#pragma unroll
    for (int e = 0; e < 4; ++e)
        if (i + e < n) d[i + e] = f2bf(s[i + e]);
}

__global__ __launch_bounds__(256) void embed_gather(const int* __restrict__ ids,
                                                    const float* __restrict__ emb,
                                                    float* __restrict__ h,
                                                    unsigned short* __restrict__ hbf) {
    int bl = blockIdx.x, t = threadIdx.x;
    int id = ids[bl];
    for (int c = t; c < HID_; c += 256) {
        float v = emb[(size_t)id * HID_ + c];
        h[(size_t)bl * HID_ + c] = v;
        hbf[(size_t)bl * HID_ + c] = f2bf(v);
    }
}

__global__ __launch_bounds__(256) void rmsnorm_k(const float* __restrict__ x,
                                                 const float* __restrict__ w,
                                                 unsigned short* __restrict__ ybf,
                                                 int cols) {
    int row = blockIdx.x, t = threadIdx.x;
    int lane = t & 31, wv = t >> 5;
    const float* xr = x + (size_t)row * cols;
    float ss = 0.0f;
    for (int c = t; c < cols; c += 256) { float v = xr[c]; ss += v * v; }
    ss += __shfl_xor(ss, 1);  ss += __shfl_xor(ss, 2);
    ss += __shfl_xor(ss, 4);  ss += __shfl_xor(ss, 8);
    ss += __shfl_xor(ss, 16);
    __shared__ float red[8];
    if (lane == 0) red[wv] = ss;
    __syncthreads();
    float tot = 0.0f;
#pragma unroll
    for (int j = 0; j < 8; ++j) tot += red[j];
    float sc = rsqrtf(tot / (float)cols + 1e-6f);
    for (int c = t; c < cols; c += 256)
        ybf[(size_t)row * cols + c] = f2bf(xr[c] * sc * w[c]);
}

__global__ __launch_bounds__(256) void ctx_gate_k(const float* __restrict__ ctx,
                                                  const float* __restrict__ nw,
                                                  const float* __restrict__ gate,
                                                  unsigned short* __restrict__ ybf) {
    int row = blockIdx.x, t = threadIdx.x;
    int lane = t & 31, wv = t >> 5;
    float v = ctx[(size_t)row * SDIM_ + t];
    float ss = v * v;
    ss += __shfl_xor(ss, 1);  ss += __shfl_xor(ss, 2);
    ss += __shfl_xor(ss, 4);  ss += __shfl_xor(ss, 8);
    ss += __shfl_xor(ss, 16);
    __shared__ float red[8];
    if (lane == 0) red[wv] = ss;
    __syncthreads();
    float tot = 0.0f;
#pragma unroll
    for (int j = 0; j < 8; ++j) tot += red[j];
    float sc = rsqrtf(tot / (float)SDIM_ + 1e-6f);
    float g = gate[(size_t)row * SDIM_ + t];
    ybf[(size_t)row * SDIM_ + t] = f2bf(v * sc * nw[t] * (g * sigmoidf_(g)));
}

__global__ __launch_bounds__(64) void rope_tables_k(float* __restrict__ cosb,
                                                    float* __restrict__ sinb) {
    int l = blockIdx.x, d = threadIdx.x;
    float f = (float)l * __powf(500000.0f, -(float)(d & 31) / 32.0f);
    cosb[l * HD_ + d] = __cosf(f);
    sinb[l * HD_ + d] = __sinf(f);
}

__global__ __launch_bounds__(256) void rope_apply_k(float* __restrict__ q,
                                                    float* __restrict__ k,
                                                    const float* __restrict__ cosb,
                                                    const float* __restrict__ sinb) {
    int idx = blockIdx.x * 256 + threadIdx.x;   // over BL_*NH_*32
    if (idx >= BL_ * NH_ * 32) return;
    int d = idx & 31;
    int h = (idx >> 5) & 7;
    int bl = idx >> 8;
    int l = bl % L_;
    size_t base = (size_t)bl * HID_ + h * HD_;
    float c1 = cosb[l * HD_ + d],      s1 = sinb[l * HD_ + d];
    float c2 = cosb[l * HD_ + 32 + d], s2 = sinb[l * HD_ + 32 + d];
    float qx1 = q[base + d], qx2 = q[base + 32 + d];
    q[base + d]      = qx1 * c1 - qx2 * s1;
    q[base + 32 + d] = qx2 * c2 + qx1 * s2;
    float kx1 = k[base + d], kx2 = k[base + 32 + d];
    k[base + d]      = kx1 * c1 - kx2 * s1;
    k[base + 32 + d] = kx2 * c2 + kx1 * s2;
}

// -------------------------------------------------------------------------
extern "C" void kernel_launch(void* const* d_in, const int* in_sizes, int n_in,
                              void* d_out, int out_size, void* d_ws, size_t ws_size,
                              hipStream_t stream) {
    const int*   ids   = (const int*)d_in[0];
    const float* S0    = (const float*)d_in[1];
    const float* embed = (const float*)d_in[2];
    const float* sr[10];
    const float* sw[10];
    for (int i = 0; i < 10; ++i) { sr[i] = (const float*)d_in[3 + i];
                                   sw[i] = (const float*)d_in[13 + i]; }
    // sr/sw order: Wk Wv Wq Wbw Wbb Waw Wab Wout nw gw
    const float* ln1 = (const float*)d_in[23];
    const float* wq  = (const float*)d_in[24];
    const float* wk  = (const float*)d_in[25];
    const float* wv  = (const float*)d_in[26];
    const float* wo  = (const float*)d_in[27];
    const float* ln2 = (const float*)d_in[28];
    const float* wg  = (const float*)d_in[29];
    const float* wu  = (const float*)d_in[30];
    const float* wd  = (const float*)d_in[31];
    const float* fnw = (const float*)d_in[32];
    float* out = (float*)d_out;

    float* W = (float*)d_ws;
    size_t o = 0;
    float* h    = W + o; o += (size_t)BL_ * HID_;
    float* qb   = W + o; o += (size_t)BL_ * HID_;
    float* kb   = W + o; o += (size_t)BL_ * HID_;
    float* vb   = W + o; o += (size_t)BL_ * HID_;
    float* up   = W + o; o += (size_t)BL_ * FFN_;
    float* kd   = W + o; o += (size_t)BL_ * SDIM_;
    float* vd   = W + o; o += (size_t)BL_ * SDIM_;
    float* qd   = W + o; o += (size_t)BL_ * SDIM_;
    float* bd   = W + o; o += (size_t)BL_ * SDIM_;
    float* ad   = W + o; o += (size_t)BL_ * SDIM_;
    float* gd   = W + o; o += (size_t)BL_ * SDIM_;
    float* ctx  = W + o; o += (size_t)BL_ * SDIM_;
    float* cosb = W + o; o += (size_t)L_ * HD_;
    float* sinb = W + o; o += (size_t)L_ * HD_;
    float* Smid = W + o; o += (size_t)B_ * SH_ * SD_ * SD_;

    // bf16 region (ushort units)
    unsigned short* U = (unsigned short*)(W + o);
    size_t uo = 0;
    auto ualloc = [&](size_t n) { unsigned short* p = U + uo; uo += (n + 7) & ~7ull; return p; };
    unsigned short* h_bf    = ualloc((size_t)BL_ * HID_);
    unsigned short* x_bf    = ualloc((size_t)BL_ * HID_);
    unsigned short* ab_bf   = ualloc((size_t)BL_ * HID_);
    unsigned short* gat_bf  = ualloc((size_t)BL_ * FFN_);
    unsigned short* ctxg_bf = ualloc((size_t)BL_ * SDIM_);

    auto cvt = [&](const float* s, size_t n) {
        unsigned short* p = ualloc(n);
        cvt_bf16_k<<<((int)n + 1023) / 1024, 256, 0, stream>>>(s, p, (int)n);
        return p;
    };
    unsigned short* wq_b = cvt(wq, (size_t)NL_ * HID_ * HID_);
    unsigned short* wk_b = cvt(wk, (size_t)NL_ * HID_ * HID_);
    unsigned short* wv_b = cvt(wv, (size_t)NL_ * HID_ * HID_);
    unsigned short* wo_b = cvt(wo, (size_t)NL_ * HID_ * HID_);
    unsigned short* wg_b = cvt(wg, (size_t)NL_ * FFN_ * HID_);
    unsigned short* wu_b = cvt(wu, (size_t)NL_ * FFN_ * HID_);
    unsigned short* wd_b = cvt(wd, (size_t)NL_ * HID_ * FFN_);
    unsigned short* em_b = cvt(embed, (size_t)VOC_ * HID_);
    unsigned short* srb[7];
    unsigned short* swb[7];
    {   // indices into sr/sw: Wk Wv Wq Wbw Waw gw Wout
        const int wi[7] = {0, 1, 2, 3, 5, 9, 7};
        for (int j = 0; j < 7; ++j) {
            size_t n = (size_t)SDIM_ * HID_;   // Wout is HID_*SDIM_ = same count
            srb[j] = cvt(sr[wi[j]], n);
            swb[j] = cvt(sw[wi[j]], n);
        }
    }

    auto gemm = [&](const unsigned short* A, const unsigned short* Bw, float* C,
                    unsigned short* Cbf, const float* extra, int M, int N, int K,
                    int mode) {
        dim3 g((N + 63) / 64, M / 128);
        gemm_wmma<<<g, 256, 0, stream>>>(A, Bw, C, Cbf, extra, M, N, K, mode);
    };

    embed_gather<<<BL_, 256, 0, stream>>>(ids, embed, h, h_bf);
    rope_tables_k<<<L_, 64, 0, stream>>>(cosb, sinb);

    auto delta = [&](const float* const* P, unsigned short* const* Pb,
                     const float* Sin, float* Sout) {
        gemm(h_bf, Pb[0], kd, nullptr, nullptr, BL_, SDIM_, HID_, 0);  // Wk
        gemm(h_bf, Pb[1], vd, nullptr, nullptr, BL_, SDIM_, HID_, 0);  // Wv
        gemm(h_bf, Pb[2], qd, nullptr, nullptr, BL_, SDIM_, HID_, 0);  // Wq
        gemm(h_bf, Pb[3], bd, nullptr, P[4],    BL_, SDIM_, HID_, 1);  // sig(Wbw+Wbb)
        gemm(h_bf, Pb[4], ad, nullptr, P[6],    BL_, SDIM_, HID_, 1);  // sig(Waw+Wab)
        gemm(h_bf, Pb[5], gd, nullptr, nullptr, BL_, SDIM_, HID_, 0);  // gw
        delta_scan<<<B_ * SH_, 32, 0, stream>>>(kd, vd, qd, bd, ad, Sin, Sout, ctx);
        ctx_gate_k<<<BL_, 256, 0, stream>>>(ctx, P[8], gd, ctxg_bf);   // nw
        gemm(ctxg_bf, Pb[6], h, h_bf, h, BL_, HID_, SDIM_, 2);         // h += ctx@Wout^T
    };

    delta(sr, srb, S0, Smid);

    for (int i = 0; i < NL_; ++i) {
        rmsnorm_k<<<BL_, 256, 0, stream>>>(h, ln1 + (size_t)i * HID_, x_bf, HID_);
        size_t wofs = (size_t)i * HID_ * HID_;
        gemm(x_bf, wq_b + wofs, qb, nullptr, nullptr, BL_, HID_, HID_, 0);
        gemm(x_bf, wk_b + wofs, kb, nullptr, nullptr, BL_, HID_, HID_, 0);
        gemm(x_bf, wv_b + wofs, vb, nullptr, nullptr, BL_, HID_, HID_, 0);
        rope_apply_k<<<(BL_ * NH_ * 32 + 255) / 256, 256, 0, stream>>>(qb, kb, cosb,
                                                                       sinb);
        attn_flash<<<dim3(L_ / 64, B_ * NH_), 128, 0, stream>>>(qb, kb, vb, ab_bf);
        gemm(ab_bf, wo_b + wofs, h, h_bf, h, BL_, HID_, HID_, 2);       // h += attn
        rmsnorm_k<<<BL_, 256, 0, stream>>>(h, ln2 + (size_t)i * HID_, x_bf, HID_);
        size_t fofs = (size_t)i * FFN_ * HID_;
        gemm(x_bf, wu_b + fofs, up, nullptr, nullptr, BL_, FFN_, HID_, 0);
        gemm(x_bf, wg_b + fofs, nullptr, gat_bf, up, BL_, FFN_, HID_, 3); // silu*up
        gemm(gat_bf, wd_b + fofs, h, h_bf, h, BL_, HID_, FFN_, 2);      // h += ffn
    }

    delta(sw, swb, Smid, out + (size_t)BL_ * VOC_);   // final S -> tail of d_out

    rmsnorm_k<<<BL_, 256, 0, stream>>>(h, fnw, x_bf, HID_);
    gemm(x_bf, em_b, out, nullptr, nullptr, BL_, VOC_, HID_, 0);        // tied logits
    (void)in_sizes; (void)n_in; (void)out_size; (void)ws_size;
}
